// MultiHeadCrossAtt_20323785245305
// MI455X (gfx1250) — matrix operand order repaired and checked by reading, not compiled, not run on hardware
//
#include <hip/hip_runtime.h>
#include <stdint.h>

typedef __attribute__((ext_vector_type(16))) _Float16 v16h;
typedef __attribute__((ext_vector_type(8)))  _Float16 v8h;
typedef __attribute__((ext_vector_type(4)))  _Float16 v4h;
typedef __attribute__((ext_vector_type(8)))  float    v8f;

// ---------------------------------------------------------------------------
// WMMA helpers (gfx1250 wave32, V_WMMA_F32_16X16X32_F16)
// ---------------------------------------------------------------------------
__device__ __forceinline__ v8f wmma16(v16h a, v16h b, v8f c) {
    return __builtin_amdgcn_wmma_f32_16x16x32_f16(
        /*neg_a=*/false, a, /*neg_b=*/false, b,
        /*c_mod=*/(short)0, c, /*reuse_a=*/false, /*reuse_b=*/false);
}

__device__ __forceinline__ v8f zero8() {
    v8f z;
#pragma unroll
    for (int i = 0; i < 8; ++i) z[i] = 0.0f;
    return z;
}

// A fragment (16x32 f16): LDS tile row-major [row][k], lda in halves (mult of 8).
__device__ __forceinline__ v16h load_a_frag(const _Float16* At, int lda,
                                            int row0, int k0, int lane) {
    int r  = row0 + (lane & 15);
    int kb = k0 + ((lane >> 4) << 3);          // +0 or +8
    const _Float16* p = At + r * lda + kb;
    v8h lo = *(const v8h*)(p);                 // K = kb .. kb+7
    v8h hi = *(const v8h*)(p + 16);            // K = kb+16 .. kb+23
    v16h a;
#pragma unroll
    for (int i = 0; i < 8; ++i) { a[i] = lo[i]; a[i + 8] = hi[i]; }
    return a;
}

// B fragment (32x16 f16): LDS tile stored TRANSPOSED [n][k], ldb in halves.
__device__ __forceinline__ v16h load_b_frag(const _Float16* Bt, int ldb,
                                            int n0, int k0, int lane) {
    int n  = n0 + (lane & 15);
    int kb = k0 + ((lane >> 4) << 3);          // +0 or +8
    const _Float16* p = Bt + n * ldb + kb;
    v8h lo = *(const v8h*)(p);
    v8h hi = *(const v8h*)(p + 16);
    v16h b;
#pragma unroll
    for (int i = 0; i < 8; ++i) { b[i] = lo[i]; b[i + 8] = hi[i]; }
    return b;
}

__device__ __forceinline__ float wave_sum(float v) {
#pragma unroll
    for (int o = 16; o > 0; o >>= 1) v += __shfl_xor(v, o, 32);
    return v;
}
__device__ __forceinline__ float wave_max(float v) {
#pragma unroll
    for (int o = 16; o > 0; o >>= 1) v = fmaxf(v, __shfl_xor(v, o, 32));
    return v;
}

// Async global->LDS copy (GLOBAL_LOAD_ASYNC_TO_LDS_B128, ASYNCcnt-tracked).
__device__ __forceinline__ void async_copy_b128(void* lds_ptr, const void* gptr) {
    uint32_t lds_off = (uint32_t)(uintptr_t)lds_ptr;   // flat addr[31:0] == LDS offset
    asm volatile("global_load_async_to_lds_b128 %0, %1, off"
                 :: "v"(lds_off), "v"(gptr) : "memory");
}
__device__ __forceinline__ void wait_async0() {
    asm volatile("s_wait_asynccnt 0x0" ::: "memory");
}

// ---------------------------------------------------------------------------
// Kernel 1: LayerNorm statistics (one wave per 1024-wide row)
// ---------------------------------------------------------------------------
__global__ __launch_bounds__(256) void ln_stats_kernel(
    const float* __restrict__ x, float* __restrict__ mu,
    float* __restrict__ rstd, int rows) {
    int w = threadIdx.x >> 5, lane = threadIdx.x & 31;
    int row = blockIdx.x * 8 + w;
    if (row >= rows) return;
    const float4* xr = (const float4*)(x + (size_t)row * 1024);
    float s = 0.f, ss = 0.f;
#pragma unroll
    for (int i = 0; i < 8; ++i) {
        float4 v = xr[lane + i * 32];
        s  += v.x + v.y + v.z + v.w;
        ss += v.x * v.x + v.y * v.y + v.z * v.z + v.w * v.w;
    }
    s = wave_sum(s); ss = wave_sum(ss);
    if (lane == 0) {
        float m   = s * (1.0f / 1024.0f);
        float var = ss * (1.0f / 1024.0f) - m * m;
        mu[row]   = m;
        rstd[row] = rsqrtf(var + 1e-5f);
    }
}

// ---------------------------------------------------------------------------
// Kernel 2: fused LN + projection GEMM:  Yh[b,h,t,64] = LN(X) @ W + bias
// 128x128 tile, BK=32, software-pipelined staging, packed LDS stores.
// ---------------------------------------------------------------------------
template <int T>
__global__ __launch_bounds__(256) void ln_project_kernel(
    const float* __restrict__ X, const float* __restrict__ mu,
    const float* __restrict__ rstd, const float* __restrict__ g,
    const float* __restrict__ beta, const float* __restrict__ W,
    const float* __restrict__ bias, _Float16* __restrict__ outh) {
    __shared__ _Float16 As[128 * 40];   // [m][k]
    __shared__ _Float16 Bs[128 * 40];   // transposed: [n][k]
    int tid = threadIdx.x, w = tid >> 5, lane = tid & 31;
    int mbase = blockIdx.x * 128, nbase = blockIdx.y * 128;
    int wm = (w & 3) * 32, wn = (w >> 2) * 64;

    // A staging map: 4 chunks per thread, chunk it -> (row am, col ac)
    int am[4], ac[4];
    float amu[4], ars[4];
#pragma unroll
    for (int it = 0; it < 4; ++it) {
        int e = tid + it * 256;
        am[it] = e >> 3; ac[it] = (e & 7) * 4;
        amu[it] = mu[mbase + am[it]];
        ars[it] = rstd[mbase + am[it]];
    }
    // B staging map: 4x4 block per thread
    int kb = (tid >> 5) * 4, nb = (tid & 31) * 4;

    v8f acc[2][4];
#pragma unroll
    for (int i = 0; i < 2; ++i)
#pragma unroll
        for (int j = 0; j < 4; ++j) acc[i][j] = zero8();

    float4 xa[4], ga[4], ba[4];
    float wbv[4][4];
    auto prefetch = [&](int k0) {
#pragma unroll
        for (int it = 0; it < 4; ++it) {
            xa[it] = *(const float4*)(X + (size_t)(mbase + am[it]) * 1024 + k0 + ac[it]);
            ga[it] = *(const float4*)(g + k0 + ac[it]);
            ba[it] = *(const float4*)(beta + k0 + ac[it]);
        }
#pragma unroll
        for (int j = 0; j < 4; ++j) {
            float4 t = *(const float4*)(W + (size_t)(k0 + kb + j) * 1024 + nbase + nb);
            wbv[j][0] = t.x; wbv[j][1] = t.y; wbv[j][2] = t.z; wbv[j][3] = t.w;
        }
    };
    prefetch(0);

    for (int k0 = 0; k0 < 1024; k0 += 32) {
        // Store staged registers to LDS (packed)
#pragma unroll
        for (int it = 0; it < 4; ++it) {
            v4h hv;
            hv[0] = (_Float16)((xa[it].x - amu[it]) * ars[it] * ga[it].x + ba[it].x);
            hv[1] = (_Float16)((xa[it].y - amu[it]) * ars[it] * ga[it].y + ba[it].y);
            hv[2] = (_Float16)((xa[it].z - amu[it]) * ars[it] * ga[it].z + ba[it].z);
            hv[3] = (_Float16)((xa[it].w - amu[it]) * ars[it] * ga[it].w + ba[it].w);
            *(v4h*)(As + am[it] * 40 + ac[it]) = hv;
        }
#pragma unroll
        for (int c = 0; c < 4; ++c) {   // register transpose of 4x4 W block
            v4h hv;
#pragma unroll
            for (int j = 0; j < 4; ++j) hv[j] = (_Float16)wbv[j][c];
            *(v4h*)(Bs + (nb + c) * 40 + kb) = hv;
        }
        __syncthreads();
        if (k0 + 32 < 1024) prefetch(k0 + 32);   // overlap next tile with WMMA
        v16h a0 = load_a_frag(As, 40, wm, 0, lane);
        v16h a1 = load_a_frag(As, 40, wm + 16, 0, lane);
#pragma unroll
        for (int j = 0; j < 4; ++j) {
            v16h bf = load_b_frag(Bs, 40, wn + j * 16, 0, lane);
            acc[0][j] = wmma16(a0, bf, acc[0][j]);
            acc[1][j] = wmma16(a1, bf, acc[1][j]);
        }
        __syncthreads();
    }
    // Epilogue: +bias, head-major f16 store (T is compile-time -> shifts)
    int msub = (lane >> 4) << 3, nsub = lane & 15;
#pragma unroll
    for (int ti = 0; ti < 2; ++ti)
#pragma unroll
        for (int tj = 0; tj < 4; ++tj) {
            int gn = nbase + wn + tj * 16 + nsub;
            float bb = bias[gn];
            int h = gn >> 6, dd = gn & 63;
#pragma unroll
            for (int r = 0; r < 8; ++r) {
                int gm = mbase + wm + ti * 16 + msub + r;
                int b = gm / T, t = gm % T;
                float val = acc[ti][tj][r] + bb;
                outh[(((size_t)(b * 16 + h)) * T + t) * 64 + dd] = (_Float16)val;
            }
        }
}

// ---------------------------------------------------------------------------
// Kernel 3: scores S[b,h,q,k] = (Qh.Kh)/8 * aux,  masked -> -1e30 sentinel
// Staging is a pure f16 copy -> async global->LDS (B128) path.
// ---------------------------------------------------------------------------
__global__ __launch_bounds__(256) void score_kernel(
    const _Float16* __restrict__ Qh, const _Float16* __restrict__ Kh,
    const float* __restrict__ aux, const int* __restrict__ mask,
    float* __restrict__ att) {
    __shared__ _Float16 Qs[128 * 72];   // [q][d]
    __shared__ _Float16 Ks[128 * 72];   // [k_tok][d] == transposed-B layout
    int tid = threadIdx.x, w = tid >> 5, lane = tid & 31;
    int qbase = blockIdx.x * 128, kbase = blockIdx.y * 128;
    int bh = blockIdx.z, b = bh >> 4;
    const _Float16* Qg = Qh + ((size_t)bh * 512 + qbase) * 64;
    const _Float16* Kg = Kh + ((size_t)bh * 2048 + kbase) * 64;
#pragma unroll
    for (int it = 0; it < 4; ++it) {       // 1024 B128 chunks per tile
        int e = tid + it * 256;
        int m = e >> 3, c = (e & 7) * 8;
        async_copy_b128(Qs + m * 72 + c, Qg + (size_t)m * 64 + c);
        async_copy_b128(Ks + m * 72 + c, Kg + (size_t)m * 64 + c);
    }
    wait_async0();
    __syncthreads();
    int wm = (w & 3) * 32, wn = (w >> 2) * 64;
    v8f acc[2][4];
#pragma unroll
    for (int i = 0; i < 2; ++i)
#pragma unroll
        for (int j = 0; j < 4; ++j) acc[i][j] = zero8();
#pragma unroll
    for (int k0 = 0; k0 < 64; k0 += 32) {
        v16h a0 = load_a_frag(Qs, 72, wm, k0, lane);
        v16h a1 = load_a_frag(Qs, 72, wm + 16, k0, lane);
#pragma unroll
        for (int j = 0; j < 4; ++j) {
            v16h bf = load_b_frag(Ks, 72, wn + j * 16, k0, lane);
            acc[0][j] = wmma16(a0, bf, acc[0][j]);
            acc[1][j] = wmma16(a1, bf, acc[1][j]);
        }
    }
    int msub = (lane >> 4) << 3, nsub = lane & 15;
#pragma unroll
    for (int ti = 0; ti < 2; ++ti)
#pragma unroll
        for (int tj = 0; tj < 4; ++tj) {
            int gk = kbase + wn + tj * 16 + nsub;
            float am = aux[b * 2048 + gk];
            int   mk = mask[b * 2048 + gk];
#pragma unroll
            for (int r = 0; r < 8; ++r) {
                int gq = qbase + wm + ti * 16 + msub + r;
                float s = acc[ti][tj][r] * 0.125f * am;
                att[((size_t)(bh * 512 + gq)) * 2048 + gk] = mk ? s : -1e30f;
            }
        }
}

// ---------------------------------------------------------------------------
// Kernel 4: row softmax over 2048 (one wave per row); masked sentinel -> 0
// ---------------------------------------------------------------------------
__global__ __launch_bounds__(256) void softmax_kernel(float* __restrict__ att) {
    int w = threadIdx.x >> 5, lane = threadIdx.x & 31;
    size_t row = (size_t)blockIdx.x * 8 + w;
    float* p = att + row * 2048;
    float vbuf[64];
    float mx = -INFINITY;
#pragma unroll
    for (int i = 0; i < 16; ++i) {
        float4 v = ((const float4*)p)[lane + i * 32];
        vbuf[i * 4 + 0] = v.x; vbuf[i * 4 + 1] = v.y;
        vbuf[i * 4 + 2] = v.z; vbuf[i * 4 + 3] = v.w;
        mx = fmaxf(mx, fmaxf(fmaxf(v.x, v.y), fmaxf(v.z, v.w)));
    }
    mx = wave_max(mx);
    float s = 0.f;
#pragma unroll
    for (int e = 0; e < 64; ++e) {
        float x  = vbuf[e];
        float ev = (x <= -1e29f) ? 0.0f : expf(x - mx);  // masked -> exactly 0
        vbuf[e] = ev;
        s += ev;
    }
    s = wave_sum(s);
    float inv = (s > 0.f) ? (1.0f / s) : 0.0f;
#pragma unroll
    for (int i = 0; i < 16; ++i) {
        float4 o;
        o.x = vbuf[i * 4 + 0] * inv; o.y = vbuf[i * 4 + 1] * inv;
        o.z = vbuf[i * 4 + 2] * inv; o.w = vbuf[i * 4 + 3] * inv;
        ((float4*)p)[lane + i * 32] = o;
    }
}

// ---------------------------------------------------------------------------
// Kernel 5: ctx[b,t,h*64+d] = att[b,h,t,:] @ Vh[b,h,:,d]   (M=512,N=64,K=2048)
// Software-pipelined; packed LDS stores; f16 4x4 register transpose for V.
// ---------------------------------------------------------------------------
__global__ __launch_bounds__(256) void ctx_kernel(
    const float* __restrict__ att, const _Float16* __restrict__ Vh,
    _Float16* __restrict__ ctx) {
    __shared__ _Float16 As[128 * 40];   // att tile [q][k] (f32->f16 on staging)
    __shared__ _Float16 Bt[64 * 40];    // V tile transposed [d][k]
    int tid = threadIdx.x, w = tid >> 5, lane = tid & 31;
    int qbase = blockIdx.x * 128;
    int bh = blockIdx.y, b = bh >> 4, h = bh & 15;
    const float*    Ag = att + ((size_t)bh * 512 + qbase) * 2048;
    const _Float16* Vg = Vh + (size_t)bh * 2048 * 64;
    int wm = (w & 3) * 32, wn = (w >> 2) * 32;
    // A map: 4 chunks; B map: 4x4 f16 block, threads < 128 only
    int am[4], ac[4];
#pragma unroll
    for (int it = 0; it < 4; ++it) {
        int e = tid + it * 256;
        am[it] = e >> 3; ac[it] = (e & 7) * 4;
    }
    int kb = (tid >> 4) * 4, db = (tid & 15) * 4;
    bool bstage = (tid < 128);

    v8f acc[2][2];
#pragma unroll
    for (int i = 0; i < 2; ++i)
#pragma unroll
        for (int j = 0; j < 2; ++j) acc[i][j] = zero8();

    float4 xa[4];
    v4h vb[4];
    auto prefetch = [&](int k0) {
#pragma unroll
        for (int it = 0; it < 4; ++it)
            xa[it] = *(const float4*)(Ag + (size_t)am[it] * 2048 + k0 + ac[it]);
        if (bstage) {
#pragma unroll
            for (int j = 0; j < 4; ++j)
                vb[j] = *(const v4h*)(Vg + (size_t)(k0 + kb + j) * 64 + db);
        }
    };
    prefetch(0);

    for (int k0 = 0; k0 < 2048; k0 += 32) {
#pragma unroll
        for (int it = 0; it < 4; ++it) {
            v4h hv;
            hv[0] = (_Float16)xa[it].x; hv[1] = (_Float16)xa[it].y;
            hv[2] = (_Float16)xa[it].z; hv[3] = (_Float16)xa[it].w;
            *(v4h*)(As + am[it] * 40 + ac[it]) = hv;
        }
        if (bstage) {
#pragma unroll
            for (int c = 0; c < 4; ++c) {   // 4x4 f16 register transpose
                v4h hv;
#pragma unroll
                for (int j = 0; j < 4; ++j) hv[j] = vb[j][c];
                *(v4h*)(Bt + (db + c) * 40 + kb) = hv;
            }
        }
        __syncthreads();
        if (k0 + 32 < 2048) prefetch(k0 + 32);
        v16h a0 = load_a_frag(As, 40, wm, 0, lane);
        v16h a1 = load_a_frag(As, 40, wm + 16, 0, lane);
#pragma unroll
        for (int j = 0; j < 2; ++j) {
            v16h bf = load_b_frag(Bt, 40, wn + j * 16, 0, lane);
            acc[0][j] = wmma16(a0, bf, acc[0][j]);
            acc[1][j] = wmma16(a1, bf, acc[1][j]);
        }
        __syncthreads();
    }
    int msub = (lane >> 4) << 3, nsub = lane & 15;
#pragma unroll
    for (int ti = 0; ti < 2; ++ti)
#pragma unroll
        for (int tj = 0; tj < 2; ++tj) {
            int d = wn + tj * 16 + nsub;
#pragma unroll
            for (int r = 0; r < 8; ++r) {
                int gq = qbase + wm + ti * 16 + msub + r;
                ctx[((size_t)(b * 512 + gq)) * 1024 + h * 64 + d] =
                    (_Float16)acc[ti][tj][r];
            }
        }
}

// ---------------------------------------------------------------------------
// Kernel 6: out = ctx @ Wo + bo   (M=2048, N=1024, K=1024), f32 output
// ---------------------------------------------------------------------------
__global__ __launch_bounds__(256) void oproj_kernel(
    const _Float16* __restrict__ A, const float* __restrict__ W,
    const float* __restrict__ bias, float* __restrict__ out) {
    __shared__ _Float16 As[128 * 40];
    __shared__ _Float16 Bs[128 * 40];   // transposed [n][k]
    int tid = threadIdx.x, w = tid >> 5, lane = tid & 31;
    int mbase = blockIdx.x * 128, nbase = blockIdx.y * 128;
    int wm = (w & 3) * 32, wn = (w >> 2) * 64;
    int am[2], ac[2];
#pragma unroll
    for (int it = 0; it < 2; ++it) {
        int e = tid + it * 256;
        am[it] = e >> 2; ac[it] = (e & 3) * 8;
    }
    int kb = (tid >> 5) * 4, nb = (tid & 31) * 4;

    v8f acc[2][4];
#pragma unroll
    for (int i = 0; i < 2; ++i)
#pragma unroll
        for (int j = 0; j < 4; ++j) acc[i][j] = zero8();

    v8h aa[2];
    float wbv[4][4];
    auto prefetch = [&](int k0) {
#pragma unroll
        for (int it = 0; it < 2; ++it)
            aa[it] = *(const v8h*)(A + (size_t)(mbase + am[it]) * 1024 + k0 + ac[it]);
#pragma unroll
        for (int j = 0; j < 4; ++j) {
            float4 t = *(const float4*)(W + (size_t)(k0 + kb + j) * 1024 + nbase + nb);
            wbv[j][0] = t.x; wbv[j][1] = t.y; wbv[j][2] = t.z; wbv[j][3] = t.w;
        }
    };
    prefetch(0);

    for (int k0 = 0; k0 < 1024; k0 += 32) {
#pragma unroll
        for (int it = 0; it < 2; ++it)
            *(v8h*)(As + am[it] * 40 + ac[it]) = aa[it];
#pragma unroll
        for (int c = 0; c < 4; ++c) {
            v4h hv;
#pragma unroll
            for (int j = 0; j < 4; ++j) hv[j] = (_Float16)wbv[j][c];
            *(v4h*)(Bs + (nb + c) * 40 + kb) = hv;
        }
        __syncthreads();
        if (k0 + 32 < 1024) prefetch(k0 + 32);
        v16h a0 = load_a_frag(As, 40, wm, 0, lane);
        v16h a1 = load_a_frag(As, 40, wm + 16, 0, lane);
#pragma unroll
        for (int j = 0; j < 4; ++j) {
            v16h bf = load_b_frag(Bs, 40, wn + j * 16, 0, lane);
            acc[0][j] = wmma16(a0, bf, acc[0][j]);
            acc[1][j] = wmma16(a1, bf, acc[1][j]);
        }
        __syncthreads();
    }
    int msub = (lane >> 4) << 3, nsub = lane & 15;
#pragma unroll
    for (int ti = 0; ti < 2; ++ti)
#pragma unroll
        for (int tj = 0; tj < 4; ++tj) {
            int gn = nbase + wn + tj * 16 + nsub;
            float bb = bias[gn];
#pragma unroll
            for (int r = 0; r < 8; ++r) {
                int gm = mbase + wm + ti * 16 + msub + r;
                out[(size_t)gm * 1024 + gn] = acc[ti][tj][r] + bb;
            }
        }
}

// ---------------------------------------------------------------------------
extern "C" void kernel_launch(void* const* d_in, const int* in_sizes, int n_in,
                              void* d_out, int out_size, void* d_ws, size_t ws_size,
                              hipStream_t stream) {
    (void)in_sizes; (void)n_in; (void)out_size; (void)ws_size;
    const float* q      = (const float*)d_in[0];
    const float* k      = (const float*)d_in[1];
    const float* v      = (const float*)d_in[2];
    const int*   mask   = (const int*)d_in[3];
    const float* aux    = (const float*)d_in[4];
    const float* ln_q_g = (const float*)d_in[5];
    const float* ln_q_b = (const float*)d_in[6];
    const float* ln_k_g = (const float*)d_in[7];
    const float* ln_k_b = (const float*)d_in[8];
    const float* ln_v_g = (const float*)d_in[9];
    const float* ln_v_b = (const float*)d_in[10];
    const float* Wq = (const float*)d_in[11];
    const float* bq = (const float*)d_in[12];
    const float* Wk = (const float*)d_in[13];
    const float* bk = (const float*)d_in[14];
    const float* Wv = (const float*)d_in[15];
    const float* bv = (const float*)d_in[16];
    const float* Wo = (const float*)d_in[17];
    const float* bo = (const float*)d_in[18];

    float* out = (float*)d_out;                         // [4,512,1024]
    float* att = out + (size_t)4 * 512 * 1024;          // [4,16,512,2048]

    _Float16* Qh  = (_Float16*)d_ws;                    // [4,16,512,64]
    _Float16* Kh  = Qh + (size_t)4 * 16 * 512 * 64;     // [4,16,2048,64]
    _Float16* Vh  = Kh + (size_t)4 * 16 * 2048 * 64;    // [4,16,2048,64]
    _Float16* Ctx = Vh + (size_t)4 * 16 * 2048 * 64;    // [4,512,1024]
    float* stats = (float*)(Ctx + (size_t)4 * 512 * 1024);
    float* muq = stats;        float* rsq = muq + 2048;
    float* muk = rsq + 2048;   float* rsk = muk + 8192;
    float* muv = rsk + 8192;   float* rsv = muv + 8192;

    ln_stats_kernel<<<2048 / 8, 256, 0, stream>>>(q, muq, rsq, 2048);
    ln_stats_kernel<<<8192 / 8, 256, 0, stream>>>(k, muk, rsk, 8192);
    ln_stats_kernel<<<8192 / 8, 256, 0, stream>>>(v, muv, rsv, 8192);

    ln_project_kernel<512><<<dim3(16, 8), 256, 0, stream>>>(q, muq, rsq, ln_q_g, ln_q_b, Wq, bq, Qh);
    ln_project_kernel<2048><<<dim3(64, 8), 256, 0, stream>>>(k, muk, rsk, ln_k_g, ln_k_b, Wk, bk, Kh);
    ln_project_kernel<2048><<<dim3(64, 8), 256, 0, stream>>>(v, muv, rsv, ln_v_g, ln_v_b, Wv, bv, Vh);

    score_kernel<<<dim3(4, 16, 64), 256, 0, stream>>>(Qh, Kh, aux, mask, att);
    softmax_kernel<<<dim3(32768 / 8), 256, 0, stream>>>(att);
    ctx_kernel<<<dim3(4, 64), 256, 0, stream>>>(att, Vh, Ctx);
    oproj_kernel<<<dim3(16, 8), 256, 0, stream>>>(Ctx, Wo, bo, out);
}
